// PagedGraniteAttention_39161511805038
// MI455X (gfx1250) — matrix operand order repaired and checked by reading, not compiled, not run on hardware
//
#include <hip/hip_runtime.h>

// ---------------------------------------------------------------------------
// Granite-8B-like paged attention block for MI455X (gfx1250, wave32, WMMA).
//
// Dims (fixed): B=4, S=1024, H=4096, NH=32, NKV=8, D=128.
// The reference's paged-cache scatter+gather is an identity permutation
// (disjoint pages), so page_table does not affect the output.
//
// Pipeline:
//   1) qraw = hidden @ wq   (bf16 out)     \
//   2) kraw = hidden @ wk   (bf16 out)      >  WMMA bf16 GEMMs
//   3) vraw = hidden @ wv   (bf16 out)     /
//   4) RoPE(qraw) -> q_l [B,NH,S,D] bf16 ; RoPE(kraw) -> k_l [B,NKV,S,D]
//   5) relayout vraw -> v_l [B,NKV,S,D]
//   6) flash attention (softcap + causal + online softmax) -> attn bf16
//        K tiles staged to LDS via the Tensor Data Mover when available.
//   7) out = attn @ wo  (f32 out to d_out)
//
// Workspace usage: 128 MB (see offsets in kernel_launch).
// ---------------------------------------------------------------------------

typedef __attribute__((ext_vector_type(16))) __bf16        v16bf;
typedef __attribute__((ext_vector_type(8)))  float         v8f;
typedef __attribute__((ext_vector_type(4)))  unsigned int  v4u;
typedef __attribute__((ext_vector_type(8)))  int           v8i;
typedef __attribute__((ext_vector_type(4)))  int           v4i;

#if defined(__has_builtin)
#if __has_builtin(__builtin_amdgcn_tensor_load_to_lds) && \
    __has_builtin(__builtin_amdgcn_s_wait_tensorcnt)
#define HAVE_TDM 1
#endif
#endif
#ifndef HAVE_TDM
#define HAVE_TDM 0
#endif

union FragBF { v16bf v; unsigned int u[8]; };

__device__ __forceinline__ v8f wmma_bf16(v16bf a, v16bf b, v8f c) {
  // D = A(16x32 bf16) * B(32x16 bf16) + C(16x16 f32)
  return __builtin_amdgcn_wmma_f32_16x16x32_bf16(
      /*neg_a=*/false, a, /*neg_b=*/false, b,
      /*c_mod=*/(short)0, c, /*reuse_a=*/false, /*reuse_b=*/false);
}

__device__ __forceinline__ v8f zero8() {
  v8f z = {0.f, 0.f, 0.f, 0.f, 0.f, 0.f, 0.f, 0.f};
  return z;
}

__device__ __forceinline__ void store_c(float* p, float v)  { *p = v; }
__device__ __forceinline__ void store_c(__bf16* p, float v) { *p = (__bf16)v; }

// Load 4 consecutive elements (f32 or bf16) and store as packed 4xbf16 (8 B).
template <typename T>
__device__ __forceinline__ void load4_to_bf16(const T* __restrict__ p,
                                              __bf16* dst) {
  if constexpr (sizeof(T) == 4) {
    const float4 v = *(const float4*)p;
    __bf16 h[4] = {(__bf16)v.x, (__bf16)v.y, (__bf16)v.z, (__bf16)v.w};
    *(unsigned long long*)dst = *(const unsigned long long*)h;
  } else {
    *(unsigned long long*)dst = *(const unsigned long long*)p;
  }
}

// ---------------------------------------------------------------------------
// Generic row-major GEMM: C[M,N] = A[M,K] * W[K,N], bf16 WMMA inside.
// Block: 128 threads (4 waves). Block tile 64x64, K step 32.
// Wave w -> 32x32 subtile at (wm = (w&1)*32, wn = (w>>1)*32).
// ---------------------------------------------------------------------------
template <typename TA, typename TW, typename TC>
__global__ __launch_bounds__(128)
void wmma_gemm_kernel(const TA* __restrict__ A, const TW* __restrict__ W,
                      TC* __restrict__ C, int M, int N, int K) {
  __shared__ __bf16 As[64][36];   // [m][k], padded to 72 B rows (8B aligned)
  __shared__ __bf16 BsT[64][36];  // [n][k], padded (W tile transposed)

  const int tid   = threadIdx.x;
  const int m0    = blockIdx.y * 64;
  const int n0    = blockIdx.x * 64;
  const int wave  = tid >> 5;
  const int lane  = tid & 31;
  const int lhalf = lane & 15;
  const int hi    = lane >> 4;
  const int wm    = (wave & 1) * 32;
  const int wn    = (wave >> 1) * 32;

  v8f acc[2][2];
#pragma unroll
  for (int i = 0; i < 2; ++i)
#pragma unroll
    for (int j = 0; j < 2; ++j) acc[i][j] = zero8();

  for (int k0 = 0; k0 < K; k0 += 32) {
    __syncthreads();
    // Stage A tile 64x32 -> bf16. 512 groups of 4 elems; 4 per thread.
#pragma unroll
    for (int i = 0; i < 4; ++i) {
      int idx = i * 128 + tid;            // 0..511
      int r = idx >> 3, kk = (idx & 7) * 4;
      load4_to_bf16(&A[(size_t)(m0 + r) * K + (k0 + kk)], &As[r][kk]);
    }
    // Stage W tile 32x64 transposed into BsT[n][k]; global float4 on n.
#pragma unroll
    for (int i = 0; i < 4; ++i) {
      int idx = i * 128 + tid;            // 0..511
      int kk = idx >> 4, nn = (idx & 15) * 4;
      if constexpr (sizeof(TW) == 4) {
        const float4 v = *(const float4*)&W[(size_t)(k0 + kk) * N + (n0 + nn)];
        BsT[nn + 0][kk] = (__bf16)v.x;
        BsT[nn + 1][kk] = (__bf16)v.y;
        BsT[nn + 2][kk] = (__bf16)v.z;
        BsT[nn + 3][kk] = (__bf16)v.w;
      } else {
        const TW* p = &W[(size_t)(k0 + kk) * N + (n0 + nn)];
        BsT[nn + 0][kk] = (__bf16)p[0];
        BsT[nn + 1][kk] = (__bf16)p[1];
        BsT[nn + 2][kk] = (__bf16)p[2];
        BsT[nn + 3][kk] = (__bf16)p[3];
      }
    }
    // Prefetch next K tile (speculative).
    if (k0 + 32 < K) {
      __builtin_prefetch(&A[(size_t)(m0 + (tid >> 1)) * K + (k0 + 32)], 0, 1);
      __builtin_prefetch(&W[(size_t)(k0 + 32 + (tid >> 2)) * N + n0], 0, 1);
    }
    __syncthreads();

    // Per-wave fragment loads from LDS.
    FragBF af[2], bfr[2];
#pragma unroll
    for (int t = 0; t < 2; ++t) {
      const int m = wm + t * 16 + lhalf;
#pragma unroll
      for (int j = 0; j < 8; ++j) {
        // ISA 16-bit A layout: element e -> K = (e<8 ? e : e+8) + 8*hi
        int kk = ((j < 4) ? 2 * j : 2 * j + 8) + hi * 8;
        af[t].u[j] = *(const unsigned int*)&As[m][kk];
      }
      const int n = wn + t * 16 + lhalf;
#pragma unroll
      for (int j = 0; j < 8; ++j) {
        // B layout: element e -> K = (e<8 ? e : e+8) + 8*hi (matches A)
        int kk = ((j < 4) ? 2 * j : 2 * j + 8) + hi * 8;
        bfr[t].u[j] = *(const unsigned int*)&BsT[n][kk];
      }
    }
#pragma unroll
    for (int i = 0; i < 2; ++i)
#pragma unroll
      for (int j = 0; j < 2; ++j)
        acc[i][j] = wmma_bf16(af[i].v, bfr[j].v, acc[i][j]);
  }

  // Store: D layout -> lane holds col (lane&15); vgpr r -> row r + 8*hi.
#pragma unroll
  for (int i = 0; i < 2; ++i) {
    const int rowb = m0 + wm + i * 16 + hi * 8;
#pragma unroll
    for (int j = 0; j < 2; ++j) {
      const int col = n0 + wn + j * 16 + lhalf;
#pragma unroll
      for (int r = 0; r < 8; ++r)
        store_c(&C[(size_t)(rowb + r) * N + col], acc[i][j][r]);
    }
  }
}

// ---------------------------------------------------------------------------
// RoPE: raw [4096 rows, nheads*128] bf16 -> out [B, nheads, S, D] bf16.
// ---------------------------------------------------------------------------
__global__ void rope_kernel(const __bf16* __restrict__ raw,
                            const float* __restrict__ cosT,
                            const float* __restrict__ sinT,
                            __bf16* __restrict__ out, int nheads) {
  int idx = blockIdx.x * blockDim.x + threadIdx.x; // (row, h, d<64)
  int d   = idx & 63;
  int h   = (idx >> 6) % nheads;
  int row = idx / (nheads * 64);
  if (row >= 4096) return;
  int s = row & 1023;  // S = 1024
  int b = row >> 10;

  size_t ib = (size_t)row * (nheads * 128) + h * 128 + d;
  float x0 = (float)raw[ib];
  float x1 = (float)raw[ib + 64];
  float c0 = cosT[s * 128 + d],      s0 = sinT[s * 128 + d];
  float c1 = cosT[s * 128 + d + 64], s1 = sinT[s * 128 + d + 64];

  size_t ob = (((size_t)b * nheads + h) * 1024 + s) * 128 + d;
  out[ob]      = (__bf16)(x0 * c0 - x1 * s0);
  out[ob + 64] = (__bf16)(x1 * c1 + x0 * s1);
}

// vraw [4096, NKV*128] -> v_l [B, NKV, S, D]  (dword copies)
__global__ void relayout_v_kernel(const __bf16* __restrict__ raw,
                                  __bf16* __restrict__ out) {
  int idx = blockIdx.x * blockDim.x + threadIdx.x; // one dword = 2 elems
  int e = idx * 2;
  if (e >= 4096 * 1024) return;
  int col = e & 1023, row = e >> 10;
  int h = col >> 7, d = col & 127;
  int s = row & 1023, b = row >> 10;
  size_t ob = (((size_t)b * 8 + h) * 1024 + s) * 128 + d;
  *(unsigned int*)(out + ob) =
      *(const unsigned int*)(raw + (size_t)row * 1024 + col);
}

// ---------------------------------------------------------------------------
// TDM: DMA one 32(keys) x 128(d) bf16 row-major tile into LDS offset 0.
// D# built per CDNA5 ISA 8.3/8.4 (2D tensor, data_size=2B, no padding).
// This toolchain exposes the 6-arg builtin:
//   (uint32x4 g0, int32x8 g1, int32x4 g2, int32x4 g3, int32x8 g4, i32 cpol)
// ---------------------------------------------------------------------------
#if HAVE_TDM
__device__ __forceinline__ void tdm_load_k_tile(const __bf16* tile_start) {
  const unsigned long long ga = (unsigned long long)(uintptr_t)tile_start;
  // Group 0: count=1 | lds_addr=0 | global_addr | type=2 ("image").
  v4u g0 = {1u, 0u, (unsigned)(ga & 0xFFFFFFFFu),
            (unsigned)((ga >> 32) & 0x1FFFFFFu) | (2u << 30)};
  // Group 1:
  //  w0: workgroup_mask=0, data_size=1 (2 bytes), no barrier/iterate/pad.
  //  tensor_dim0=128 (bits 79:48), tensor_dim1=1024 (bits 111:80),
  //  tile_dim0=128 (127:112), tile_dim1=32 (143:128), tile_dim2=0,
  //  tensor_dim0_stride=128 (207:160), tensor_dim1_stride=0.
  v8i g1 = {(int)0x00010000, (int)(128u << 16), (int)(1024u << 16),
            (int)(128u << 16), 32, 128, 0, 0};
  v4i z4 = {0, 0, 0, 0};               // groups 2/3 unused for 2D tiles
  v8i z8 = {0, 0, 0, 0, 0, 0, 0, 0};   // extra group (unused)
  __builtin_amdgcn_tensor_load_to_lds(g0, g1, z4, z4, z8, /*cpol=*/0);
}
#endif

// ---------------------------------------------------------------------------
// Flash attention with softcap + causal mask + online softmax.
// Grid: (B*NH, S/64). Block: 128 thr = 4 waves; wave owns 16 query rows.
// Dynamic LDS layout (byte offsets):
//   Ks  @ 0      : [32][128] bf16 (8192 B)  <- TDM target
//   VsT @ 8192   : [128][34] bf16 (8704 B)
//   Ps  @ 16896  : [4][16][34] bf16 (4352 B)
// ---------------------------------------------------------------------------
#define ATTN_SMEM_BYTES (8192 + 8704 + 4352)

__global__ __launch_bounds__(128)
void attn_kernel(const __bf16* __restrict__ Qg, const __bf16* __restrict__ Kg,
                 const __bf16* __restrict__ Vg, __bf16* __restrict__ Og) {
  constexpr float kScale   = 0.08838834764831845f;
  constexpr float kSoftcap = 50.0f;

  extern __shared__ __align__(16) char smem[];
  __bf16* Ks  = (__bf16*)(smem);           // [key][d], stride 128
  __bf16* VsT = (__bf16*)(smem + 8192);    // [d][key], stride 34
  __bf16* Ps  = (__bf16*)(smem + 16896);   // [wave][qrow][key], 16x34 per wave

  const int tid   = threadIdx.x;
  const int wave  = tid >> 5;
  const int lane  = tid & 31;
  const int lhalf = lane & 15;
  const int hi    = lane >> 4;

  const int bh  = blockIdx.x;       // b*32 + h
  const int b   = bh >> 5;
  const int h   = bh & 31;
  const int kvh = h >> 2;           // GQA group = 4
  const int qblk = blockIdx.y * 64;
  const int q0   = qblk + wave * 16;

  const __bf16* Q = Qg + (size_t)bh * 1024 * 128;
  const __bf16* K = Kg + ((size_t)(b * 8 + kvh)) * 1024 * 128;
  const __bf16* V = Vg + ((size_t)(b * 8 + kvh)) * 1024 * 128;

  // Preload Q fragments for 4 D-chunks of 32 (reused across all key chunks).
  FragBF qf[4];
#pragma unroll
  for (int c = 0; c < 4; ++c)
#pragma unroll
    for (int j = 0; j < 8; ++j) {
      int d = 32 * c + ((j < 4) ? 2 * j : 2 * j + 8) + hi * 8;
      qf[c].u[j] = *(const unsigned int*)(Q + (size_t)(q0 + lhalf) * 128 + d);
    }

  v8f acc[8];  // 16 q-rows x 128 d (8 col-tiles)
#pragma unroll
  for (int t = 0; t < 8; ++t) acc[t] = zero8();
  float mrow[8], lrow[8];
#pragma unroll
  for (int r = 0; r < 8; ++r) { mrow[r] = -1e30f; lrow[r] = 0.f; }

  const int kend = qblk + 64;  // keys [0, kend) cover all queries of block
  for (int kc = 0; kc < kend; kc += 32) {
    __syncthreads();
#if HAVE_TDM
    // Wave 0 DMAs the K tile into LDS via the Tensor Data Mover.
    if (wave == 0) tdm_load_k_tile(K + (size_t)kc * 128);
#else
    // Manual K staging: 512 x 16-byte copies (4 per thread).
#pragma unroll
    for (int i = 0; i < 4; ++i) {
      int idx = i * 128 + tid;           // 0..511
      int key = idx >> 4, db = (idx & 15) * 8;
      *(v4u*)&Ks[key * 128 + db] =
          *(const v4u*)(K + (size_t)(kc + key) * 128 + db);
    }
#endif
    // Stage V chunk transposed [d][key] (16-byte reads along d).
#pragma unroll
    for (int i = 0; i < 4; ++i) {
      int idx = i * 128 + tid;           // 0..511 groups of 8 elems
      int key = idx >> 4, db = (idx & 15) * 8;
      const __bf16* src = V + (size_t)(kc + key) * 128 + db;
#pragma unroll
      for (int e = 0; e < 8; ++e) VsT[(db + e) * 34 + key] = src[e];
    }
#if HAVE_TDM
    if (wave == 0) __builtin_amdgcn_s_wait_tensorcnt(0);
#endif
    __syncthreads();

    // Scores: two 16(key)-col tiles, reduce over D in 4 chunks of 32.
    v8f sc[2];
    sc[0] = zero8(); sc[1] = zero8();
#pragma unroll
    for (int nt = 0; nt < 2; ++nt) {
#pragma unroll
      for (int c = 0; c < 4; ++c) {
        FragBF kf;
#pragma unroll
        for (int j = 0; j < 8; ++j) {
          int d = 32 * c + ((j < 4) ? 2 * j : 2 * j + 8) + hi * 8;
          kf.u[j] = *(const unsigned int*)&Ks[(nt * 16 + lhalf) * 128 + d];
        }
        sc[nt] = wmma_bf16(qf[c].v, kf.v, sc[nt]);
      }
    }

    // Scale -> softcap -> causal mask; per-row block max.
    float p[2][8], mblk[8];
#pragma unroll
    for (int r = 0; r < 8; ++r) mblk[r] = -1e30f;
#pragma unroll
    for (int nt = 0; nt < 2; ++nt) {
      const int key = kc + nt * 16 + lhalf;
#pragma unroll
      for (int r = 0; r < 8; ++r) {
        const int qrow = q0 + r + 8 * hi;
        float x = sc[nt][r] * kScale;
        x = kSoftcap * tanhf(x * (1.0f / kSoftcap));
        x = (key <= qrow) ? x : -1e30f;
        p[nt][r] = x;
        mblk[r] = fmaxf(mblk[r], x);
      }
    }
    // Row max across the 16 lanes of each half-wave.
#pragma unroll
    for (int r = 0; r < 8; ++r) {
      float v = mblk[r];
      v = fmaxf(v, __shfl_xor(v, 1, 32));
      v = fmaxf(v, __shfl_xor(v, 2, 32));
      v = fmaxf(v, __shfl_xor(v, 4, 32));
      v = fmaxf(v, __shfl_xor(v, 8, 32));
      mblk[r] = v;
    }
    // Online softmax update.
    float rs[8];
#pragma unroll
    for (int r = 0; r < 8; ++r) {
      float mn   = fmaxf(mrow[r], mblk[r]);
      float corr = __expf(mrow[r] - mn);
      mrow[r] = mn;
      lrow[r] *= corr;
#pragma unroll
      for (int t = 0; t < 8; ++t) acc[t][r] *= corr;
      float sum = 0.f;
#pragma unroll
      for (int nt = 0; nt < 2; ++nt) {
        float e = __expf(p[nt][r] - mn);
        p[nt][r] = e;
        sum += e;
      }
      rs[r] = sum;
    }
#pragma unroll
    for (int r = 0; r < 8; ++r) {
      float v = rs[r];
      v += __shfl_xor(v, 1, 32);
      v += __shfl_xor(v, 2, 32);
      v += __shfl_xor(v, 4, 32);
      v += __shfl_xor(v, 8, 32);
      lrow[r] += v;
    }
    // Repack P (C layout) into A layout through LDS.
    __bf16* Pw = Ps + wave * (16 * 34);
#pragma unroll
    for (int nt = 0; nt < 2; ++nt)
#pragma unroll
      for (int r = 0; r < 8; ++r)
        Pw[(r + 8 * hi) * 34 + nt * 16 + lhalf] = (__bf16)p[nt][r];
    __syncthreads();

    FragBF pf;
#pragma unroll
    for (int j = 0; j < 8; ++j) {
      int kk = ((j < 4) ? 2 * j : 2 * j + 8) + hi * 8;
      pf.u[j] = *(const unsigned int*)&Pw[lhalf * 34 + kk];
    }
    // PV: accumulate 16x128 output (8 col-tiles of 16).
#pragma unroll
    for (int t = 0; t < 8; ++t) {
      FragBF vf;
#pragma unroll
      for (int j = 0; j < 8; ++j)
        vf.u[j] =
            *(const unsigned int*)&VsT[(t * 16 + lhalf) * 34 + ((j < 4) ? 2 * j : 2 * j + 8) + hi * 8];
      acc[t] = wmma_bf16(pf.v, vf.v, acc[t]);
    }
  }

  // Normalize and store to attn [b*S + q, h*128 + d] (row-major 4096x4096).
#pragma unroll
  for (int t = 0; t < 8; ++t) {
#pragma unroll
    for (int r = 0; r < 8; ++r) {
      const int qrow = q0 + r + 8 * hi;
      float o = acc[t][r] / lrow[r];
      Og[(size_t)(b * 1024 + qrow) * 4096 + h * 128 + t * 16 + lhalf] =
          (__bf16)o;
    }
  }
}

// ---------------------------------------------------------------------------
// Host launcher.
// ---------------------------------------------------------------------------
extern "C" void kernel_launch(void* const* d_in, const int* in_sizes, int n_in,
                              void* d_out, int out_size, void* d_ws,
                              size_t ws_size, hipStream_t stream) {
  (void)in_sizes; (void)n_in; (void)out_size; (void)ws_size;

  const float* hidden = (const float*)d_in[0];  // [4,1024,4096]
  const float* wq     = (const float*)d_in[1];  // [4096,4096]
  const float* wk     = (const float*)d_in[2];  // [4096,1024]
  const float* wv     = (const float*)d_in[3];  // [4096,1024]
  const float* wo     = (const float*)d_in[4];  // [4096,4096]
  const float* cosT   = (const float*)d_in[5];  // [1024,128]
  const float* sinT   = (const float*)d_in[6];  // [1024,128]
  // d_in[7] = page_table: scatter+gather through disjoint pages is identity.

  // Workspace layout (bytes); total = 128 MB.
  char* ws = (char*)d_ws;
  __bf16* qraw = (__bf16*)(ws + 0);          // [4096,4096]  33,554,432 B
  __bf16* kraw = (__bf16*)(ws + 33554432);   // [4096,1024]   8,388,608 B
  __bf16* vraw = (__bf16*)(ws + 41943040);   // [4096,1024]   8,388,608 B
  __bf16* q_l  = (__bf16*)(ws + 50331648);   // [4,32,1024,128]
  __bf16* k_l  = (__bf16*)(ws + 83886080);   // [4,8,1024,128]
  __bf16* v_l  = (__bf16*)(ws + 92274688);   // [4,8,1024,128]
  __bf16* attn = (__bf16*)(ws + 100663296);  // [4096,4096]

  const dim3 blk128(128);

  // 1-3) QKV projections (f32 in -> bf16 out).
  wmma_gemm_kernel<float, float, __bf16>
      <<<dim3(4096 / 64, 4096 / 64), blk128, 0, stream>>>(
          hidden, wq, qraw, 4096, 4096, 4096);
  wmma_gemm_kernel<float, float, __bf16>
      <<<dim3(1024 / 64, 4096 / 64), blk128, 0, stream>>>(
          hidden, wk, kraw, 4096, 1024, 4096);
  wmma_gemm_kernel<float, float, __bf16>
      <<<dim3(1024 / 64, 4096 / 64), blk128, 0, stream>>>(
          hidden, wv, vraw, 4096, 1024, 4096);

  // 4) RoPE + head-major relayout.
  rope_kernel<<<(4096 * 32 * 64) / 256, 256, 0, stream>>>(qraw, cosT, sinT,
                                                          q_l, 32);
  rope_kernel<<<(4096 * 8 * 64) / 256, 256, 0, stream>>>(kraw, cosT, sinT,
                                                         k_l, 8);
  // 5) V relayout.
  relayout_v_kernel<<<(4096 * 1024 / 2) / 256, 256, 0, stream>>>(vraw, v_l);

  // 6) Attention (dynamic LDS so the TDM lds_addr offsets are known).
  attn_kernel<<<dim3(4 * 32, 1024 / 64), blk128, ATTN_SMEM_BYTES, stream>>>(
      q_l, k_l, v_l, attn);

  // 7) Output projection (bf16 x f32 -> f32 to d_out).
  wmma_gemm_kernel<__bf16, float, float>
      <<<dim3(4096 / 64, 4096 / 64), blk128, 0, stream>>>(
          attn, wo, (float*)d_out, 4096, 4096, 4096);
}
